// ScaledDotProductAttention_9723805958658
// MI455X (gfx1250) — matrix operand order repaired and checked by reading, not compiled, not run on hardware
//
#include <hip/hip_runtime.h>
#include <math.h>

// MI455X / gfx1250: wave32, WMMA 16x16x32 bf16 (f32 accum), TDM async copy.
typedef __attribute__((ext_vector_type(16))) __bf16 bf16x16;
typedef __attribute__((ext_vector_type(8)))  float  f32x8;
typedef __attribute__((ext_vector_type(4)))  unsigned int u32x4;
typedef __attribute__((ext_vector_type(8)))  int   i32x8;
typedef __attribute__((ext_vector_type(4)))  int   i32x4;

#if defined(__has_builtin)
#if __has_builtin(__builtin_amdgcn_tensor_load_to_lds) && \
    __has_builtin(__builtin_amdgcn_s_wait_tensorcnt)
#define USE_TDM 1
#endif
#endif
#ifndef USE_TDM
#define USE_TDM 0
#endif

static constexpr int BB = 4, HH = 16, SS = 1024, DD = 64;
static constexpr int AROW_STRIDE = 33;     // 32 DW + 1 DW TDM pad -> no bank conflicts

__device__ __forceinline__ __bf16 f32_to_bf16(float f) {
    unsigned u = __builtin_bit_cast(unsigned, f);
    unsigned r = u + 0x7FFFu + ((u >> 16) & 1u);   // round-to-nearest-even
    unsigned short h = (unsigned short)(r >> 16);
    return __builtin_bit_cast(__bf16, h);
}

// A-matrix (16x32 bf16) per-lane element -> k offset.  Lane holds row M=lane%16.
//   k = half*16 + e     (contiguous order, matches B fragments)
__device__ __forceinline__ int a_klocal(int e, int half) {
    return (half << 4) + e;
}

#if USE_TDM
// TDM: async-load a 128x32 f32 tile (row stride 1024 elems) into LDS,
// padding +1 DWORD per 32-DWORD row (D# per CDNA5 ISA 8.3/8.4).
__device__ __forceinline__ void tdm_load_tile_128x32(const float* gsrc,
                                                     unsigned lds_off) {
    const unsigned long long ga = (unsigned long long)(uintptr_t)gsrc;
    u32x4 g0;
    g0[0] = 1u;                                            // count=1, no gather
    g0[1] = lds_off;                                       // lds_addr (bytes)
    g0[2] = (unsigned)ga;                                  // global_addr[31:0]
    g0[3] = (unsigned)((ga >> 32) & 0x01FFFFFFull)         // global_addr[56:32]
            | (2u << 30);                                  // type = 2 (image)
    i32x8 g1;
    g1[0] = (int)((2u << 16)       // data_size = 4B
                  | (1u << 20)     // pad_enable
                  | (4u << 22));   // pad_interval: 32 DWORDs
                                   // pad_amount[31:25]=0 -> 1 DWORD
    g1[1] = (int)(32u << 16);      // tensor_dim0[15:0] = 32   (bits 79:48)
    g1[2] = (int)(128u << 16);     // tensor_dim1[15:0] = 128  (bits 111:80)
    g1[3] = (int)(32u << 16);      // tile_dim0 = 32           (bits 127:112)
    g1[4] = 128;                   // tile_dim1 = 128, tile_dim2 = 0
    g1[5] = 1024;                  // tensor_dim0_stride = 1024 elems
    g1[6] = 0;                     // stride hi / tensor_dim1_stride lo
    g1[7] = 0;
    const i32x4 z4 = {0, 0, 0, 0};
    const i32x8 z8 = {0, 0, 0, 0, 0, 0, 0, 0};
    // amdgpu-toolchain (clang-23) 6-arg form: (g0, g1, g2, g3, extra, cpol)
    __builtin_amdgcn_tensor_load_to_lds(g0, g1, z4, z4, z8, 0);
}
#endif

// ---------------------------------------------------------------------------
// C[row, d] = sum_t A[row, t] * Bsrc[t, d]   per (b,h) slice.
// A: [S x S] f32 (lda = S), Bsrc: [S x 64] f32, Cout: [S x 64] bf16 or f32.
// Grid: (S/128, B*H); 256 threads = 8 waves; wave owns 16 rows x 64 cols.
// A tile (128x32) arrives via TDM into padded LDS; B tile staged transposed
// as bf16 so B fragments are contiguous 32-byte LDS reads.
// ---------------------------------------------------------------------------
template <bool OUT_BF16, bool NT_A>
__launch_bounds__(256)
__global__ void gemm_k1024_n64(const float* __restrict__ A,
                               const float* __restrict__ Bsrc,
                               void* __restrict__ Cout) {
    __shared__ __bf16 ldsBt[DD * 32];           // B tile transposed [d][t], 4 KB
#if USE_TDM
    __shared__ float ldsA[128 * AROW_STRIDE + 8];  // 128x32 f32, padded rows
#endif

    const int tid  = threadIdx.x;
    const int lane = tid & 31, wave = tid >> 5;
    const int half = lane >> 4, ln = lane & 15;
    const int bh   = blockIdx.y;
    const int rowBase = blockIdx.x * 128 + wave * 16;

    const size_t aOff = (size_t)bh * SS * SS;
    const size_t bOff = (size_t)bh * SS * DD;
#if USE_TDM
    const float* aBlock = &A[aOff + (size_t)(blockIdx.x * 128) * SS];
    const unsigned ldsAoff = (unsigned)(uintptr_t)&ldsA[0];
#else
    const float* aRowPtr = &A[aOff + (size_t)(rowBase + ln) * SS];
#endif

    f32x8 acc[4] = {};                          // n-tiles d0 = 0,16,32,48

    for (int kk = 0; kk < SS; kk += 32) {
#if USE_TDM
        if (wave == 0)                           // one descriptor per WG
            tdm_load_tile_128x32(aBlock + kk, ldsAoff);
#endif
        // ---- stage B tile (32 x 64 f32 -> bf16, transposed) into LDS ----
        {
            const int tl = tid >> 3;             // 0..31  (t_local)
            const int dg = (tid & 7) * 8;        // 0..56  (8 cols each)
            const float4* p =
                (const float4*)&Bsrc[bOff + (size_t)(kk + tl) * DD + dg];
            float4 f0 = p[0], f1 = p[1];
            ldsBt[(dg + 0) * 32 + tl] = f32_to_bf16(f0.x);
            ldsBt[(dg + 1) * 32 + tl] = f32_to_bf16(f0.y);
            ldsBt[(dg + 2) * 32 + tl] = f32_to_bf16(f0.z);
            ldsBt[(dg + 3) * 32 + tl] = f32_to_bf16(f0.w);
            ldsBt[(dg + 4) * 32 + tl] = f32_to_bf16(f1.x);
            ldsBt[(dg + 5) * 32 + tl] = f32_to_bf16(f1.y);
            ldsBt[(dg + 6) * 32 + tl] = f32_to_bf16(f1.z);
            ldsBt[(dg + 7) * 32 + tl] = f32_to_bf16(f1.w);
        }
#if USE_TDM
        if (wave == 0)
            __builtin_amdgcn_s_wait_tensorcnt(0);
#endif
        __syncthreads();

        // ---- A fragment (16x32 bf16) ----
        bf16x16 afrag;
#if USE_TDM
        {
            const float* arow = &ldsA[(wave * 16 + ln) * AROW_STRIDE];
#pragma unroll
            for (int e = 0; e < 16; ++e)
                afrag[e] = f32_to_bf16(arow[a_klocal(e, half)]);
        }
#else
#pragma unroll
        for (int e = 0; e < 16; ++e) {
            const float* p = &aRowPtr[kk + a_klocal(e, half)];
            float v = NT_A ? __builtin_nontemporal_load(p) : *p;
            afrag[e] = f32_to_bf16(v);
        }
        if (kk + 32 < SS)
            __builtin_prefetch(&aRowPtr[kk + 32 + (half << 3)], 0, 0);
#endif

        // ---- 4 WMMAs: B fragment = 16 contiguous bf16 per lane from LDS ----
#pragma unroll
        for (int n = 0; n < 4; ++n) {
            const bf16x16* bp =
                (const bf16x16*)&ldsBt[(n * 16 + ln) * 32 + half * 16];
            acc[n] = __builtin_amdgcn_wmma_f32_16x16x32_bf16(
                false, afrag, false, *bp, (short)0, acc[n], false, false);
        }
        __syncthreads();
    }

    // ---- epilogue: C layout = row (r + half*8), col (n*16 + ln) ----
    const int outRow = rowBase + half * 8;
    if (OUT_BF16) {
        __bf16* C = (__bf16*)Cout + bOff;
#pragma unroll
        for (int n = 0; n < 4; ++n)
#pragma unroll
            for (int r = 0; r < 8; ++r)
                C[(size_t)(outRow + r) * DD + n * 16 + ln] =
                    f32_to_bf16(acc[n][r]);
    } else {
        float* C = (float*)Cout + bOff;
#pragma unroll
        for (int n = 0; n < 4; ++n)
#pragma unroll
            for (int r = 0; r < 8; ++r)
                C[(size_t)(outRow + r) * DD + n * 16 + ln] = acc[n][r];
    }
}

// ---------------------------------------------------------------------------
// attn[s, u] = softmax_u( (1/8) * Q[s,:] . KM[u,:] )   per (b,h).
// One wave per block, 16 query rows, 64 KB LDS score strip (5 WGs / WGP).
// ---------------------------------------------------------------------------
__launch_bounds__(32)
__global__ void scores_softmax(const float* __restrict__ Q,
                               const void* __restrict__ KMv,
                               float* __restrict__ attn) {
    extern __shared__ float sc[];              // 16 x 1024 f32
    const __bf16* KM = (const __bf16*)KMv;

    const int lane = threadIdx.x & 31;
    const int half = lane >> 4, ln = lane & 15;
    const int bh   = blockIdx.y;
    const int sBase = blockIdx.x * 16;

    // Q fragments (k = 0..31, 32..63); fold scale 1/sqrt(64) = 0.125 in.
    const size_t qOff = ((size_t)bh * SS + sBase + ln) * DD;
    bf16x16 aq[2];
#pragma unroll
    for (int f = 0; f < 2; ++f)
#pragma unroll
        for (int e = 0; e < 16; ++e)
            aq[f][e] = f32_to_bf16(Q[qOff + f * 32 + a_klocal(e, half)] * 0.125f);

    // scores2 = Q . KM^T : B fragment rows are contiguous KM rows (bf16).
    const size_t kmOff = (size_t)bh * SS * DD;
    for (int u0 = 0; u0 < SS; u0 += 16) {
        const __bf16* kb = &KM[kmOff + (size_t)(u0 + ln) * DD];
        bf16x16 b0 = *(const bf16x16*)&kb[half * 16];        // d = 0..31
        bf16x16 b1 = *(const bf16x16*)&kb[32 + half * 16];   // d = 32..63
        f32x8 c = {};
        c = __builtin_amdgcn_wmma_f32_16x16x32_bf16(false, aq[0], false, b0,
                                                    (short)0, c, false, false);
        c = __builtin_amdgcn_wmma_f32_16x16x32_bf16(false, aq[1], false, b1,
                                                    (short)0, c, false, false);
        float* dst = &sc[(half * 8) * SS + u0 + ln];
#pragma unroll
        for (int r = 0; r < 8; ++r) dst[r * SS] = c[r];
    }

    // row-wise softmax (max-subtract), write attn f32.
    for (int r = 0; r < 16; ++r) {
        float* row = &sc[r * SS];
        float m = -__builtin_inff();
        for (int c = lane; c < SS; c += 32) m = fmaxf(m, row[c]);
#pragma unroll
        for (int off = 16; off > 0; off >>= 1)
            m = fmaxf(m, __shfl_xor(m, off, 32));
        float sum = 0.f;
        for (int c = lane; c < SS; c += 32) {
            float e = __expf(row[c] - m);
            row[c] = e;
            sum += e;
        }
#pragma unroll
        for (int off = 16; off > 0; off >>= 1) sum += __shfl_xor(sum, off, 32);
        const float inv = 1.0f / sum;
        float* out = &attn[((size_t)bh * SS + sBase + r) * SS];
        for (int c = lane; c < SS; c += 32) out[c] = row[c] * inv;
    }
}

extern "C" void kernel_launch(void* const* d_in, const int* in_sizes, int n_in,
                              void* d_out, int out_size, void* d_ws,
                              size_t ws_size, hipStream_t stream) {
    (void)in_sizes; (void)n_in; (void)out_size; (void)ws_size;
    const float* Q    = (const float*)d_in[0];
    const float* K    = (const float*)d_in[1];
    const float* V    = (const float*)d_in[2];
    const float* Mask = (const float*)d_in[3];   // d_in[4] = d_k (64), folded.

    float* ctx  = (float*)d_out;                           // [B,H,S,D]
    float* attn = ctx + (size_t)BB * HH * SS * DD;         // [B,H,S,S]
    void*  KMbf = d_ws;                                    // 8 MiB bf16 [B,H,S,D]

    const dim3 gGemm(SS / 128, BB * HH);
    // KM = mask @ K   (mask streamed once, TDM tiles into LDS)
    gemm_k1024_n64<true, true><<<gGemm, 256, 0, stream>>>(Mask, K, KMbf);
    // attn = softmax( Q @ KM^T / sqrt(d) )
    scores_softmax<<<dim3(SS / 16, BB * HH), 32,
                     16 * SS * sizeof(float), stream>>>(Q, KMbf, attn);
    // ctx = attn @ V
    gemm_k1024_n64<false, false><<<gGemm, 256, 0, stream>>>(attn, V, (void*)ctx);
}
